// GNNCASimple_4209067950360
// MI455X (gfx1250) — matrix operand order repaired and checked by reading, not compiled, not run on hardware
//
#include <hip/hip_runtime.h>
#include <hip/hip_bf16.h>
#include <stdint.h>

// ---- problem constants (from reference) ----
#define N_NODES 100000
#define F_STATE 64
#define HIDDEN  256
#define N_EDGES 1600000
#define STEPS   4

typedef __attribute__((ext_vector_type(16))) _Float16 v16h;
typedef __attribute__((ext_vector_type(8)))  _Float16 v8h;
typedef __attribute__((ext_vector_type(8)))  float    v8f;
typedef __attribute__((ext_vector_type(2)))  int      v2i;

// ---- CDNA5 async global->LDS path (guarded; falls back to sync staging) ----
#if defined(__AMDGCN__) && defined(__has_builtin)
#if __has_builtin(__builtin_amdgcn_global_load_async_to_lds_b64)
#define HAVE_ASYNC_LDS 1
#endif
#endif
#ifndef HAVE_ASYNC_LDS
#define HAVE_ASYNC_LDS 0
#endif

// exact pointer types per the builtin prototype leaked by clang's diagnostic:
//   param0: v2i in address space 1 (global), param1: v2i in address space 3 (LDS)
typedef v2i __attribute__((address_space(1)))* as1_v2i;
typedef v2i __attribute__((address_space(3)))* as3_v2i;

template <int N>
__device__ __forceinline__ void wait_asynccnt() {
#if defined(__AMDGCN__) && defined(__has_builtin) && __has_builtin(__builtin_amdgcn_s_wait_asynccnt)
    __builtin_amdgcn_s_wait_asynccnt(N);
#else
    asm volatile("s_wait_asynccnt %0" ::"i"(N) : "memory");
#endif
}

__device__ __forceinline__ v16h cat8(v8h lo, v8h hi) {
    return __builtin_shufflevector(lo, hi, 0,1,2,3,4,5,6,7,8,9,10,11,12,13,14,15);
}

// =====================================================================
// Weight packing: W[K,N] f32 row-major -> CDNA5 WMMA B-fragment tiles.
// Tile (kt,nt): lanes 0-15 hold K=kt*32+{0..7,16..23} (N=lane), lanes 16-31 hold
// K=kt*32+{8..15,24..31} (N=lane-16); each lane's 16 halves contiguous (32B).
// =====================================================================
__global__ void k_pack_b(const float* __restrict__ W, _Float16* __restrict__ Wp,
                         int K, int N) {
    int t = blockIdx.x * blockDim.x + threadIdx.x;
    int NT = N >> 4;
    int total = (K >> 5) * NT * 32;
    if (t >= total) return;
    int lane = t & 31;
    int tile = t >> 5;
    int nt = tile % NT;
    int kt = tile / NT;
    int n  = (nt << 4) + (lane & 15);
    int kb = (kt << 5) + ((lane >> 4) << 3);
    _Float16* dst = Wp + (size_t)tile * 512 + lane * 16;
#pragma unroll
    for (int j = 0; j < 16; ++j)
        dst[j] = (_Float16)W[(size_t)(kb + j + ((j >> 3) << 3)) * N + n];
}

// =====================================================================
// Pre MLP:  p = relu(h @ W_pre + b_pre)    h:[N,64] f32
// 32-row blocks, 8 waves x (two 16x16 tiles each), N covered by grid.y.
// =====================================================================
__global__ __launch_bounds__(256) void k_pre(const float* __restrict__ h,
                                             const _Float16* __restrict__ Wp,
                                             const float* __restrict__ bias,
                                             float* __restrict__ p32,
                                             _Float16* __restrict__ p16) {
    __shared__ _Float16 As[32 * 32];
    const int m0   = blockIdx.x << 5;
    const int wave = threadIdx.x >> 5;
    const int lane = threadIdx.x & 31;
    const int nt   = blockIdx.y * 8 + wave;
    const int n0   = nt << 4;
    const int NT   = HIDDEN / 16;
    v8f c0 = {}, c1 = {};

    for (int kt = 0; kt < F_STATE / 32; ++kt) {
        {   // stage 32x32 chunk, f32->f16: 4 elems/thread
            int sidx = threadIdx.x << 2;
            int sr = sidx >> 5, sc = sidx & 31;
            float4 v = *(const float4*)(h + (size_t)(m0 + sr) * F_STATE + (kt << 5) + sc);
            As[sidx]     = (_Float16)v.x;
            As[sidx + 1] = (_Float16)v.y;
            As[sidx + 2] = (_Float16)v.z;
            As[sidx + 3] = (_Float16)v.w;
        }
        __syncthreads();
        const int ar = lane & 15;
        const int kb = (lane >> 4) << 3;
        v16h b  = *(const v16h*)(Wp + (size_t)(kt * NT + nt) * 512 + lane * 16);
        v16h a0 = cat8(*(const v8h*)(As + ar * 32 + kb),
                       *(const v8h*)(As + ar * 32 + kb + 16));
        v16h a1 = cat8(*(const v8h*)(As + (16 + ar) * 32 + kb),
                       *(const v8h*)(As + (16 + ar) * 32 + kb + 16));
        c0 = __builtin_amdgcn_wmma_f32_16x16x32_f16(false, a0, false, b, (short)0, c0, false, false);
        c1 = __builtin_amdgcn_wmma_f32_16x16x32_f16(false, a1, false, b, (short)0, c1, false, false);
        __syncthreads();
    }
    const int col = n0 + (lane & 15);
    const float bv = bias[col];
    const int r0 = m0 + ((lane >> 4) << 3);
#pragma unroll
    for (int r = 0; r < 8; ++r) {
        float v = c0[r] + bv; v = v > 0.f ? v : 0.f;
        size_t o = (size_t)(r0 + r) * HIDDEN + col;
        p32[o] = v; p16[o] = (_Float16)v;
        float w = c1[r] + bv; w = w > 0.f ? w : 0.f;
        size_t o2 = (size_t)(r0 + 16 + r) * HIDDEN + col;
        p32[o2] = w; p16[o2] = (_Float16)w;
    }
}

// =====================================================================
// Conv dense:  m = relu(p @ W_conv + b_conv)    p16:[N,256] f16
// Triple-buffered async global->LDS pipeline, one barrier per K-step.
// =====================================================================
__global__ __launch_bounds__(256) void k_conv(const _Float16* __restrict__ p16,
                                              const _Float16* __restrict__ Wp,
                                              const float* __restrict__ bias,
                                              _Float16* __restrict__ m16) {
    __shared__ _Float16 As[3][32 * 32];
    const int m0   = blockIdx.x << 5;
    const int wave = threadIdx.x >> 5;
    const int lane = threadIdx.x & 31;
    const int nt   = blockIdx.y * 8 + wave;
    const int n0   = nt << 4;
    const int NT   = HIDDEN / 16;
    const int KT   = HIDDEN / 32;   // 8
    v8f c0 = {}, c1 = {};

    // staging geometry: 1024 halves per chunk, 4 halves (8B) per thread
    const int sidx = threadIdx.x << 2;
    const int sr = sidx >> 5, sc = sidx & 31;
    const _Float16* grow = p16 + (size_t)(m0 + sr) * HIDDEN + sc;

#if HAVE_ASYNC_LDS
#define ASYNC_COPY(g, l) \
    __builtin_amdgcn_global_load_async_to_lds_b64((as1_v2i)(g), (as3_v2i)(l), 0, 0)
#else
#define ASYNC_COPY(g, l) ((void)0)
#endif
#define STAGE(ktv, bufv)                                                             \
    do {                                                                             \
        if (HAVE_ASYNC_LDS) {                                                        \
            /* one async VMEM instr per wave: 32 lanes x 8B -> LDS */                \
            ASYNC_COPY(grow + ((ktv) << 5), &As[(bufv)][sidx]);                      \
        } else {                                                                     \
            *(uint64_t*)&As[(bufv)][sidx] = *(const uint64_t*)(grow + ((ktv) << 5)); \
        }                                                                            \
    } while (0)

    STAGE(0, 0);
    for (int kt = 0; kt < KT; ++kt) {
        if (kt + 1 < KT) {
            STAGE(kt + 1, (kt + 1) % 3);
#if HAVE_ASYNC_LDS
            wait_asynccnt<1>();   // stage(kt) complete (this wave)
#endif
        } else {
#if HAVE_ASYNC_LDS
            wait_asynccnt<0>();
#endif
        }
        __syncthreads();          // all waves' stage(kt) visible (waits DScnt too)
        const _Float16* buf = As[kt % 3];
        const int ar = lane & 15;
        const int kb = (lane >> 4) << 3;
        v16h b  = *(const v16h*)(Wp + (size_t)(kt * NT + nt) * 512 + lane * 16);
        v16h a0 = cat8(*(const v8h*)(buf + ar * 32 + kb),
                       *(const v8h*)(buf + ar * 32 + kb + 16));
        v16h a1 = cat8(*(const v8h*)(buf + (16 + ar) * 32 + kb),
                       *(const v8h*)(buf + (16 + ar) * 32 + kb + 16));
        c0 = __builtin_amdgcn_wmma_f32_16x16x32_f16(false, a0, false, b, (short)0, c0, false, false);
        c1 = __builtin_amdgcn_wmma_f32_16x16x32_f16(false, a1, false, b, (short)0, c1, false, false);
    }
#undef STAGE
#undef ASYNC_COPY

    const int col = n0 + (lane & 15);
    const float bv = bias[col];
    const int r0 = m0 + ((lane >> 4) << 3);
#pragma unroll
    for (int r = 0; r < 8; ++r) {
        float v = c0[r] + bv; v = v > 0.f ? v : 0.f;
        m16[(size_t)(r0 + r) * HIDDEN + col] = (_Float16)v;
        float w = c1[r] + bv; w = w > 0.f ? w : 0.f;
        m16[(size_t)(r0 + 16 + r) * HIDDEN + col] = (_Float16)w;
    }
}

// =====================================================================
// Edge scatter:  z[dst] += m[src]   (segment_sum)
// 32 threads/edge, 8 halves (b128 gather) each; f32 atomics into
// L2-resident z. m(51MB)+z(102MB) < 192MB L2.
// =====================================================================
__global__ __launch_bounds__(256) void k_scatter(const int* __restrict__ dst,
                                                 const int* __restrict__ src,
                                                 const _Float16* __restrict__ m16,
                                                 float* __restrict__ z) {
    long long t = (long long)blockIdx.x * blockDim.x + threadIdx.x;
    int e = (int)(t >> 5);
    if (e >= N_EDGES) return;
    int cc = ((int)t & 31) << 3;
    int s = src[e], d = dst[e];
    v8h mv = *(const v8h*)(m16 + (size_t)s * HIDDEN + cc);
    float* zp = z + (size_t)d * HIDDEN + cc;
#pragma unroll
    for (int i = 0; i < 8; ++i)
        atomicAdd(zp + i, (float)mv[i]);
}

// =====================================================================
// Post MLP:  out = concat([z, p]) @ W_post + b_post   (linear)
// K=512: k-tiles 0..7 from z (f32->f16), 8..15 from p16. N=64 -> 4 waves.
// 32-row blocks, two accumulators per wave.
// =====================================================================
__global__ __launch_bounds__(128) void k_post(const float* __restrict__ z,
                                              const _Float16* __restrict__ p16,
                                              const _Float16* __restrict__ Wp,
                                              const float* __restrict__ bias,
                                              float* __restrict__ out) {
    __shared__ _Float16 As[32 * 32];
    const int m0   = blockIdx.x << 5;
    const int wave = threadIdx.x >> 5;
    const int lane = threadIdx.x & 31;
    const int n0   = wave << 4;              // N tiles 0..3
    const int NT   = F_STATE / 16;           // 4
    v8f c0 = {}, c1 = {};

    for (int kt = 0; kt < (2 * HIDDEN) / 32; ++kt) {
        {   // stage 1024 halves with 128 threads: 8 halves/thread
            int sidx = threadIdx.x << 3;
            int sr = sidx >> 5, sc = sidx & 31;
            if (kt < 8) {
                const float* zr = z + (size_t)(m0 + sr) * HIDDEN + (kt << 5) + sc;
                float4 v0 = *(const float4*)(zr);
                float4 v1 = *(const float4*)(zr + 4);
                As[sidx]     = (_Float16)v0.x; As[sidx + 1] = (_Float16)v0.y;
                As[sidx + 2] = (_Float16)v0.z; As[sidx + 3] = (_Float16)v0.w;
                As[sidx + 4] = (_Float16)v1.x; As[sidx + 5] = (_Float16)v1.y;
                As[sidx + 6] = (_Float16)v1.z; As[sidx + 7] = (_Float16)v1.w;
            } else {
                *(uint4*)(As + sidx) =
                    *(const uint4*)(p16 + (size_t)(m0 + sr) * HIDDEN + ((kt - 8) << 5) + sc);
            }
        }
        __syncthreads();
        const int ar = lane & 15;
        const int kb = (lane >> 4) << 3;
        v16h b  = *(const v16h*)(Wp + (size_t)(kt * NT + wave) * 512 + lane * 16);
        v16h a0 = cat8(*(const v8h*)(As + ar * 32 + kb),
                       *(const v8h*)(As + ar * 32 + kb + 16));
        v16h a1 = cat8(*(const v8h*)(As + (16 + ar) * 32 + kb),
                       *(const v8h*)(As + (16 + ar) * 32 + kb + 16));
        c0 = __builtin_amdgcn_wmma_f32_16x16x32_f16(false, a0, false, b, (short)0, c0, false, false);
        c1 = __builtin_amdgcn_wmma_f32_16x16x32_f16(false, a1, false, b, (short)0, c1, false, false);
        __syncthreads();
    }
    const int col = n0 + (lane & 15);
    const float bv = bias[col];
    const int r0 = m0 + ((lane >> 4) << 3);
#pragma unroll
    for (int r = 0; r < 8; ++r) {
        out[(size_t)(r0 + r) * F_STATE + col]      = c0[r] + bv;
        out[(size_t)(r0 + 16 + r) * F_STATE + col] = c1[r] + bv;
    }
}

// =====================================================================
extern "C" void kernel_launch(void* const* d_in, const int* in_sizes, int n_in,
                              void* d_out, int out_size, void* d_ws, size_t ws_size,
                              hipStream_t stream) {
    (void)in_sizes; (void)n_in; (void)out_size; (void)ws_size;
    const float* x      = (const float*)d_in[0];
    const int*   ei     = (const int*)d_in[1];
    const float* W_pre  = (const float*)d_in[2];
    const float* b_pre  = (const float*)d_in[3];
    const float* W_conv = (const float*)d_in[4];
    const float* b_conv = (const float*)d_in[5];
    const float* W_post = (const float*)d_in[6];
    const float* b_post = (const float*)d_in[7];
    const int* dst = ei;              // edge_index[0] : aggregation targets
    const int* src = ei + N_EDGES;    // edge_index[1] : message sources
    float* out = (float*)d_out;

    char* ws = (char*)d_ws;
    size_t off = 0;
    auto carve = [&](size_t bytes) -> void* {
        void* p = ws + off;
        off = (off + bytes + 255) & ~(size_t)255;
        return p;
    };
    float*    p32 = (float*)   carve((size_t)N_NODES * HIDDEN * 4);
    _Float16* p16 = (_Float16*)carve((size_t)N_NODES * HIDDEN * 2);
    _Float16* m16 = (_Float16*)carve((size_t)N_NODES * HIDDEN * 2);
    float*    zb  = (float*)   carve((size_t)N_NODES * HIDDEN * 4);
    _Float16* Wp_pre  = (_Float16*)carve((size_t)F_STATE * HIDDEN * 2);
    _Float16* Wp_conv = (_Float16*)carve((size_t)HIDDEN * HIDDEN * 2);
    _Float16* Wp_post = (_Float16*)carve((size_t)2 * HIDDEN * F_STATE * 2);

    {
        int t0 = (F_STATE / 32) * (HIDDEN / 16) * 32;
        int t1 = (HIDDEN / 32) * (HIDDEN / 16) * 32;
        int t2 = ((2 * HIDDEN) / 32) * (F_STATE / 16) * 32;
        k_pack_b<<<(t0 + 255) / 256, 256, 0, stream>>>(W_pre,  Wp_pre,  F_STATE,    HIDDEN);
        k_pack_b<<<(t1 + 255) / 256, 256, 0, stream>>>(W_conv, Wp_conv, HIDDEN,     HIDDEN);
        k_pack_b<<<(t2 + 255) / 256, 256, 0, stream>>>(W_post, Wp_post, 2 * HIDDEN, F_STATE);
    }

    const int MT2 = N_NODES / 32;   // 3125 double-M-tile blocks
    for (int s = 0; s < STEPS; ++s) {
        const float* h = (s == 0) ? x : (const float*)out;
        k_pre <<<dim3(MT2, HIDDEN / 128), 256, 0, stream>>>(h, Wp_pre, b_pre, p32, p16);
        k_conv<<<dim3(MT2, HIDDEN / 128), 256, 0, stream>>>(p16, Wp_conv, b_conv, m16);
        (void)hipMemsetAsync(zb, 0, (size_t)N_NODES * HIDDEN * 4, stream);
        {
            long long nthr = (long long)N_EDGES * 32;
            k_scatter<<<(unsigned)((nthr + 255) / 256), 256, 0, stream>>>(dst, src, m16, zb);
        }
        k_post<<<dim3(MT2, 1), 128, 0, stream>>>(zb, p16, Wp_post, b_post, out);
    }
}
